// S2VT_47914655154531
// MI455X (gfx1250) — matrix operand order repaired and checked by reading, not compiled, not run on hardware
//
#include <hip/hip_runtime.h>
#include <hip/hip_bf16.h>
#include <stdint.h>

// Model dims
#define BATCH   8
#define STEPS   80
#define T2      160
#define FRAME   4096
#define HID     1000
#define H4      4000
#define VEMB    500
#define WEMB    500
#define VOCAB   32000

typedef __attribute__((ext_vector_type(16))) __bf16 bf16x16;
typedef __attribute__((ext_vector_type(8)))  float  floatx8;

// native f32 -> bf16 convert (backend picks the hw cvt op)
__device__ __forceinline__ unsigned short f2bf(float x) {
  union { __bf16 b; unsigned short u; } v;
  v.b = (__bf16)x;
  return v.u;
}

// ---------------------------------------------------------------------------
// Generic GEMM: Cout[m,n] = sum_k A[m,k]*Bsrc[n,k] + bias[n] + Cin[m,n]
//   A: row-major [M,K] (f32 or bf16), Bsrc: row-major [N,K] (f32 or bf16)
//   256 threads = 8 waves; each wave: 16(M)x64(N) slab via 4 WMMAs; block
//   tile 16x512, K chunk 32. permT>0: row remap (b-major -> t-major emit).
// ---------------------------------------------------------------------------
template<bool ABF16, bool BBF16>
__global__ __launch_bounds__(256) void gemm_wmma_bf16(
    const void* __restrict__ Av, int lda,
    const void* __restrict__ Bv, int ldb,
    const float* __restrict__ bias,
    const float* __restrict__ Cin, int ldcin,
    float* __restrict__ Cout, int ldc,
    int M, int N, int K, int permT)
{
  __shared__ unsigned short As[16 * 32];
  __shared__ unsigned short Bs[512 * 32];

  const int tid  = threadIdx.x;
  const int lane = tid & 31;
  const int wave = tid >> 5;
  const int mtile  = blockIdx.y * 16;
  const int nblock = blockIdx.x * 512;

  floatx8 acc[4];
#pragma unroll
  for (int s = 0; s < 4; ++s)
#pragma unroll
    for (int r = 0; r < 8; ++r) acc[s][r] = 0.0f;

  unsigned* As32 = (unsigned*)As;
  unsigned* Bs32 = (unsigned*)Bs;

  for (int k0 = 0; k0 < K; k0 += 32) {
    __syncthreads();
    // ---- stage A tile (16x32) ----
    if (tid < 128) {
      int r = tid >> 3, c = (tid & 7) * 4;
      int gm = mtile + r, gk = k0 + c;
      unsigned short e0 = 0, e1 = 0, e2 = 0, e3 = 0;
      if (gm < M) {
        if (ABF16) {
          const unsigned short* ap = (const unsigned short*)Av + (size_t)gm * lda + gk;
          if (gk + 3 < K) { e0 = ap[0]; e1 = ap[1]; e2 = ap[2]; e3 = ap[3]; }
          else {
            if (gk     < K) e0 = ap[0];
            if (gk + 1 < K) e1 = ap[1];
            if (gk + 2 < K) e2 = ap[2];
          }
        } else {
          const float* ap = (const float*)Av + (size_t)gm * lda + gk;
          if (gk + 3 < K) {
            float4 f = *(const float4*)ap;
            e0 = f2bf(f.x); e1 = f2bf(f.y); e2 = f2bf(f.z); e3 = f2bf(f.w);
          } else {
            if (gk     < K) e0 = f2bf(ap[0]);
            if (gk + 1 < K) e1 = f2bf(ap[1]);
            if (gk + 2 < K) e2 = f2bf(ap[2]);
          }
        }
      }
      int o = (r * 32 + c) >> 1;
      As32[o]     = (unsigned)e0 | ((unsigned)e1 << 16);
      As32[o + 1] = (unsigned)e2 | ((unsigned)e3 << 16);
    }
    // ---- stage B tile (512x32) ----
    for (int i = tid; i < 512 * 8; i += 256) {
      int r = i >> 3, c = (i & 7) * 4;
      int gn = nblock + r, gk = k0 + c;
      unsigned short e0 = 0, e1 = 0, e2 = 0, e3 = 0;
      if (gn < N) {
        if (BBF16) {
          const unsigned short* bp = (const unsigned short*)Bv + (size_t)gn * ldb + gk;
          if (gk + 3 < K) { e0 = bp[0]; e1 = bp[1]; e2 = bp[2]; e3 = bp[3]; }
          else {
            if (gk     < K) e0 = bp[0];
            if (gk + 1 < K) e1 = bp[1];
            if (gk + 2 < K) e2 = bp[2];
          }
        } else {
          const float* bp = (const float*)Bv + (size_t)gn * ldb + gk;
          if (gk + 3 < K) {
            float4 f = *(const float4*)bp;
            e0 = f2bf(f.x); e1 = f2bf(f.y); e2 = f2bf(f.z); e3 = f2bf(f.w);
          } else {
            if (gk     < K) e0 = f2bf(bp[0]);
            if (gk + 1 < K) e1 = f2bf(bp[1]);
            if (gk + 2 < K) e2 = f2bf(bp[2]);
          }
        }
      }
      int o = (r * 32 + c) >> 1;
      Bs32[o]     = (unsigned)e0 | ((unsigned)e1 << 16);
      Bs32[o + 1] = (unsigned)e2 | ((unsigned)e3 << 16);
    }
    __syncthreads();

    // ---- A fragment (16x32 bf16, documented CDNA5 layout) ----
    union { bf16x16 v; unsigned u[8]; } af;
    const int lh = lane >> 4, arow = lane & 15;
#pragma unroll
    for (int j = 0; j < 8; ++j) {
      int kb = ((j & 4) ? 16 : 0) + lh * 8 + (j & 3) * 2;
      af.u[j] = As32[arow * 16 + (kb >> 1)];
    }
    // ---- B fragments + WMMA ----
    const int bcol = lane & 15, khalf = lane >> 4;
#pragma unroll
    for (int s = 0; s < 4; ++s) {
      union { bf16x16 v; unsigned u[8]; } bfr;
      int n0 = wave * 64 + s * 16;
#pragma unroll
      for (int j = 0; j < 8; ++j)
        bfr.u[j] = Bs32[(n0 + bcol) * 16 + ((j & 4) ? 8 : 0) + khalf * 4 + (j & 3)];
      acc[s] = __builtin_amdgcn_wmma_f32_16x16x32_bf16(
          false, af.v, false, bfr.v, (short)0, acc[s], false, false);
    }
  }

  const int lh = lane >> 4, lcol = lane & 15;
#pragma unroll
  for (int s = 0; s < 4; ++s) {
    int gn = nblock + wave * 64 + s * 16 + lcol;
    if (gn >= N) continue;
    float bsum = bias ? bias[gn] : 0.0f;
#pragma unroll
    for (int r = 0; r < 8; ++r) {
      int gm = mtile + r + lh * 8;
      if (gm >= M) continue;
      float v = acc[s][r] + bsum;
      if (Cin) v += Cin[(size_t)gm * ldcin + gn];
      int orow = gm;
      if (permT) orow = (gm % permT) * (M / permT) + (gm / permT);
      Cout[(size_t)orow * ldc + gn] = v;
    }
  }
}

// ---------------------------------------------------------------------------
// Fused LSTM step: gates GEMM + cell update in one kernel.
//   Each wave's 4 sub-tiles sit at gate offsets {0,1,2,3}*1000 + j, so after
//   the K-loop the wave holds i/f/g/o for its (m, j) slab and applies the
//   cell update in-register. 64 threads = 2 waves, 32 j-columns per block,
//   grid = 32 blocks over j in [0,1000).
//   A split: k <  1000 -> hA_lo[m*2000 + k]   (h1)
//            k >= 1000 -> hA_hi[m*2000 + k]   (h2), ping-pong buffers.
//   W: bf16 [4000, K] row-major. pre_t: f32 [8,4000]. c: f32 [8,1000].
//   hout: bf16, row stride 2000 (pre-offset by caller). hseq nullable.
// ---------------------------------------------------------------------------
__global__ __launch_bounds__(64) void lstm_gemm_cell(
    const unsigned short* __restrict__ hA_lo,
    const unsigned short* __restrict__ hA_hi,
    const unsigned short* __restrict__ W, int K,
    const float* __restrict__ pre_t,
    float* __restrict__ c,
    unsigned short* __restrict__ hout,
    unsigned short* __restrict__ hseq, int tt)
{
  __shared__ unsigned short As[16 * 32];
  __shared__ unsigned short Bs[128 * 32];

  const int tid  = threadIdx.x;
  const int lane = tid & 31;
  const int wave = tid >> 5;            // 0..1
  const int jblock = blockIdx.x * 32;   // 32 j-columns per block

  floatx8 acc[4];
#pragma unroll
  for (int s = 0; s < 4; ++s)
#pragma unroll
    for (int r = 0; r < 8; ++r) acc[s][r] = 0.0f;

  unsigned* As32 = (unsigned*)As;
  unsigned* Bs32 = (unsigned*)Bs;

  for (int k0 = 0; k0 < K; k0 += 32) {
    __syncthreads();
    // ---- stage A tile (16x32): rows 0..7 = batch, rows 8..15 zero ----
    for (int i = tid; i < 128; i += 64) {
      int r = i >> 3, cc = (i & 7) * 4;
      int gk = k0 + cc;
      unsigned short e0 = 0, e1 = 0, e2 = 0, e3 = 0;
      if (r < BATCH) {
        const unsigned short* ap =
            ((gk < HID) ? hA_lo : hA_hi) + (size_t)r * (2 * HID) + gk;
        if (gk + 3 < K) { e0 = ap[0]; e1 = ap[1]; e2 = ap[2]; e3 = ap[3]; }
        else {
          if (gk     < K) e0 = ap[0];
          if (gk + 1 < K) e1 = ap[1];
          if (gk + 2 < K) e2 = ap[2];
        }
      }
      int o = (r * 32 + cc) >> 1;
      As32[o]     = (unsigned)e0 | ((unsigned)e1 << 16);
      As32[o + 1] = (unsigned)e2 | ((unsigned)e3 << 16);
    }
    // ---- stage B tile: 128 rows = 2 waves x 4 gates x 16 j ----
    for (int i = tid; i < 128 * 8; i += 64) {
      int r = i >> 3, cc = (i & 7) * 4;
      int wv = r >> 6, s = (r >> 4) & 3, col = r & 15;
      int jg = jblock + wv * 16 + col;
      int gn = s * HID + jg;               // weight row (gates stacked i,f,g,o)
      int gk = k0 + cc;
      unsigned short e0 = 0, e1 = 0, e2 = 0, e3 = 0;
      if (jg < HID) {
        const unsigned short* bp = W + (size_t)gn * K + gk;
        if (gk + 3 < K) { e0 = bp[0]; e1 = bp[1]; e2 = bp[2]; e3 = bp[3]; }
        else {
          if (gk     < K) e0 = bp[0];
          if (gk + 1 < K) e1 = bp[1];
          if (gk + 2 < K) e2 = bp[2];
        }
      }
      int o = (r * 32 + cc) >> 1;
      Bs32[o]     = (unsigned)e0 | ((unsigned)e1 << 16);
      Bs32[o + 1] = (unsigned)e2 | ((unsigned)e3 << 16);
    }
    __syncthreads();

    union { bf16x16 v; unsigned u[8]; } af;
    const int lh = lane >> 4, arow = lane & 15;
#pragma unroll
    for (int j = 0; j < 8; ++j) {
      int kb = ((j & 4) ? 16 : 0) + lh * 8 + (j & 3) * 2;
      af.u[j] = As32[arow * 16 + (kb >> 1)];
    }
    const int bcol = lane & 15, khalf = lane >> 4;
#pragma unroll
    for (int s = 0; s < 4; ++s) {
      union { bf16x16 v; unsigned u[8]; } bfr;
      int n0 = wave * 64 + s * 16;
#pragma unroll
      for (int j = 0; j < 8; ++j)
        bfr.u[j] = Bs32[(n0 + bcol) * 16 + ((j & 4) ? 8 : 0) + khalf * 4 + (j & 3)];
      acc[s] = __builtin_amdgcn_wmma_f32_16x16x32_bf16(
          false, af.v, false, bfr.v, (short)0, acc[s], false, false);
    }
  }

  // ---- fused cell epilogue: lanes 0-15 hold m = r (rows 0..7 valid) ----
  const int lh = lane >> 4, col = lane & 15;
  int j = jblock + wave * 16 + col;
  if (lh == 0 && j < HID) {
#pragma unroll
    for (int r = 0; r < BATCH; ++r) {
      const float* p = pre_t + (size_t)r * H4;
      float gi = acc[0][r] + p[j];
      float gf = acc[1][r] + p[HID + j];
      float gg = acc[2][r] + p[2 * HID + j];
      float go = acc[3][r] + p[3 * HID + j];
      float si = 1.0f / (1.0f + __expf(-gi));
      float sf = 1.0f / (1.0f + __expf(-gf));
      float so = 1.0f / (1.0f + __expf(-go));
      float cn = sf * c[r * HID + j] + si * tanhf(gg);
      c[r * HID + j] = cn;
      unsigned short hb = f2bf(so * tanhf(cn));
      hout[(size_t)r * (2 * HID) + j] = hb;
      if (hseq) hseq[((size_t)r * STEPS + tt) * HID + j] = hb;
    }
  }
}

// ---------------------------------------------------------------------------
// Elementwise helpers
// ---------------------------------------------------------------------------
__global__ __launch_bounds__(256) void argmax_kernel(
    const float* __restrict__ logits, float* __restrict__ out)
{
  __shared__ float sv[256];
  __shared__ int   si[256];
  int row = blockIdx.x;
  const float* p = logits + (size_t)row * VOCAB;
  float best = -3.4e38f;
  int bidx = 0x7fffffff;
  for (int j = threadIdx.x; j < VOCAB; j += 256) {
    float v = p[j];
    if (v > best || (v == best && j < bidx)) { best = v; bidx = j; }
  }
  sv[threadIdx.x] = best; si[threadIdx.x] = bidx;
  __syncthreads();
  for (int s = 128; s > 0; s >>= 1) {
    if (threadIdx.x < s) {
      float v = sv[threadIdx.x + s]; int ix = si[threadIdx.x + s];
      if (v > sv[threadIdx.x] || (v == sv[threadIdx.x] && ix < si[threadIdx.x])) {
        sv[threadIdx.x] = v; si[threadIdx.x] = ix;
      }
    }
    __syncthreads();
  }
  if (threadIdx.x == 0) out[row] = (float)si[0];
}

__global__ void zero_f32_kernel(float* p, int n) {
  int i = blockIdx.x * 256 + threadIdx.x;
  if (i < n) p[i] = 0.0f;
}
__global__ void zero_u16_kernel(unsigned short* p, int n) {
  int i = blockIdx.x * 256 + threadIdx.x;
  if (i < n) p[i] = 0;
}
__global__ void add_vec_kernel(const float* a, const float* b, float* o, int n) {
  int i = blockIdx.x * 256 + threadIdx.x;
  if (i < n) o[i] = a[i] + b[i];
}
__global__ void conv_f32_bf16_kernel(const float* s, unsigned short* d, int n) {
  int i = blockIdx.x * 256 + threadIdx.x;
  if (i < n) d[i] = f2bf(s[i]);
}
// comb2[n, 0:1000] = W_ih2[n, 0:1000]; comb2[n, 1000:2000] = W_hh2[n, :]
__global__ void build_comb2_kernel(const float* Wih2, const float* Whh2,
                                   unsigned short* comb) {
  int i = blockIdx.x * 256 + threadIdx.x;
  if (i >= H4 * 2 * HID) return;
  int r = i / (2 * HID), c = i % (2 * HID);
  float v = (c < HID) ? Wih2[(size_t)r * (WEMB + HID) + c]
                      : Whh2[(size_t)r * HID + (c - HID)];
  comb[i] = f2bf(v);
}
// word vectors in (t,b)-major order: row = t*8 + b
__global__ void gather_words_kernel(const int* cap, const float* Wemb, float* outv) {
  int i = blockIdx.x * 256 + threadIdx.x;
  if (i >= STEPS * BATCH * WEMB) return;
  int row = i / WEMB, j = i % WEMB;
  int t = row >> 3, b = row & 7;
  int tok = cap[b * STEPS + t];
  outv[i] = Wemb[(size_t)tok * WEMB + j];
}
__global__ void fill_bias_rows_kernel(float* dst, const float* bias, int rows, int ncols) {
  int i = blockIdx.x * 256 + threadIdx.x;
  if (i < rows * ncols) dst[i] = bias[i % ncols];
}

// ---------------------------------------------------------------------------
extern "C" void kernel_launch(void* const* d_in, const int* in_sizes, int n_in,
                              void* d_out, int out_size, void* d_ws, size_t ws_size,
                              hipStream_t stream) {
  const float* video = (const float*)d_in[0];
  const int*   cap   = (const int*)d_in[1];
  const float* W_emb = (const float*)d_in[2];
  const float* W_vid = (const float*)d_in[3];
  const float* b_vid = (const float*)d_in[4];
  const float* W_ih1 = (const float*)d_in[5];
  const float* W_hh1 = (const float*)d_in[6];
  const float* b_ih1 = (const float*)d_in[7];
  const float* b_hh1 = (const float*)d_in[8];
  const float* W_ih2 = (const float*)d_in[9];
  const float* W_hh2 = (const float*)d_in[10];
  const float* b_ih2 = (const float*)d_in[11];
  const float* b_hh2 = (const float*)d_in[12];
  const float* W_out = (const float*)d_in[13];
  const float* b_out = (const float*)d_in[14];
  float* out = (float*)d_out;        // [0..639]=argmax as float, [640..]=logits

  // workspace bump allocator (~72 MB total)
  char* ws = (char*)d_ws;
  size_t off = 0;
  auto alloc = [&](size_t bytes) -> void* {
    void* p = ws + off;
    off = (off + bytes + 255) & ~(size_t)255;
    return p;
  };
  float* ve    = (float*)alloc((size_t)STEPS * BATCH * VEMB * 4);    // (t,b)-major
  float* wv    = (float*)alloc((size_t)STEPS * BATCH * WEMB * 4);    // (t,b)-major
  float* pre1  = (float*)alloc((size_t)T2 * BATCH * H4 * 4);
  float* pre2  = (float*)alloc((size_t)T2 * BATCH * H4 * 4);
  float* c1    = (float*)alloc((size_t)BATCH * HID * 4);
  float* c2    = (float*)alloc((size_t)BATCH * HID * 4);
  unsigned short* hbufA = (unsigned short*)alloc((size_t)BATCH * 2 * HID * 2);
  unsigned short* hbufB = (unsigned short*)alloc((size_t)BATCH * 2 * HID * 2);
  unsigned short* hseq  = (unsigned short*)alloc((size_t)BATCH * STEPS * HID * 2);
  float* bc1   = (float*)alloc((size_t)H4 * 4);
  float* bc2   = (float*)alloc((size_t)H4 * 4);
  unsigned short* Whh1b = (unsigned short*)alloc((size_t)H4 * HID * 2);
  unsigned short* comb2 = (unsigned short*)alloc((size_t)H4 * 2 * HID * 2);

  const int TB = STEPS * BATCH;  // 640

  // init state
  zero_f32_kernel<<<(BATCH * HID + 255) / 256, 256, 0, stream>>>(c1, BATCH * HID);
  zero_f32_kernel<<<(BATCH * HID + 255) / 256, 256, 0, stream>>>(c2, BATCH * HID);
  zero_u16_kernel<<<(BATCH * 2 * HID + 255) / 256, 256, 0, stream>>>(hbufA, BATCH * 2 * HID);
  zero_u16_kernel<<<(BATCH * 2 * HID + 255) / 256, 256, 0, stream>>>(hbufB, BATCH * 2 * HID);
  // combined biases
  add_vec_kernel<<<(H4 + 255) / 256, 256, 0, stream>>>(b_ih1, b_hh1, bc1, H4);
  add_vec_kernel<<<(H4 + 255) / 256, 256, 0, stream>>>(b_ih2, b_hh2, bc2, H4);
  // bf16 recurrent weights
  conv_f32_bf16_kernel<<<(H4 * HID + 255) / 256, 256, 0, stream>>>(W_hh1, Whh1b, H4 * HID);
  build_comb2_kernel<<<(H4 * 2 * HID + 255) / 256, 256, 0, stream>>>(W_ih2, W_hh2, comb2);
  // word embedding gather (t-major)
  gather_words_kernel<<<(TB * WEMB + 255) / 256, 256, 0, stream>>>(cap, W_emb, wv);
  // pre-activation bias fills for the padded halves
  fill_bias_rows_kernel<<<(TB * H4 + 255) / 256, 256, 0, stream>>>(
      pre1 + (size_t)TB * H4, bc1, TB, H4);                       // t in [80,160)
  fill_bias_rows_kernel<<<(TB * H4 + 255) / 256, 256, 0, stream>>>(
      pre2, bc2, TB, H4);                                         // t in [0,80)

  // video embeds: [640,4096] x W_vid[N=500,K=4096] -> ve (t-major via permT=80)
  gemm_wmma_bf16<false, false><<<dim3(1, 40), 256, 0, stream>>>(
      video, FRAME, W_vid, FRAME, b_vid, nullptr, 0, ve, VEMB,
      TB, VEMB, FRAME, STEPS);
  // pre1[0:640] = ve @ W_ih1^T + (b_ih1+b_hh1)
  gemm_wmma_bf16<false, false><<<dim3(8, 40), 256, 0, stream>>>(
      ve, VEMB, W_ih1, VEMB, bc1, nullptr, 0, pre1, H4,
      TB, H4, VEMB, 0);
  // pre2[640:1280] = wv @ W_ih2[:,1000:1500]^T + (b_ih2+b_hh2)
  gemm_wmma_bf16<false, false><<<dim3(8, 40), 256, 0, stream>>>(
      wv, WEMB, W_ih2 + HID, WEMB + HID, bc2, nullptr, 0,
      pre2 + (size_t)TB * H4, H4, TB, H4, WEMB, 0);

  // sequential recurrence: 160 steps x 2 fused GEMM+cell kernels,
  // ping-pong h buffers to avoid intra-kernel read/write hazards.
  for (int t = 0; t < T2; ++t) {
    unsigned short* rd = (t & 1) ? hbufB : hbufA;   // [h1_{t-1} | h2_{t-1}]
    unsigned short* wr = (t & 1) ? hbufA : hbufB;
    // LSTM1: gates = pre1[t] + h1_{t-1} @ W_hh1^T ; writes h1_t -> wr[:,0:1000]
    lstm_gemm_cell<<<32, 64, 0, stream>>>(
        rd, rd, Whh1b, HID,
        pre1 + (size_t)t * BATCH * H4, c1, wr, nullptr, 0);
    // LSTM2: gates = pre2[t] + [h1_t | h2_{t-1}] @ [W_ih2[:,:H] | W_hh2]^T
    //        reads h1_t from wr (k<1000), h2_{t-1} from rd (k>=1000);
    //        writes h2_t -> wr[:,1000:2000]
    lstm_gemm_cell<<<32, 64, 0, stream>>>(
        wr, rd, comb2, 2 * HID,
        pre2 + (size_t)t * BATCH * H4, c2, wr + HID,
        (t >= STEPS) ? hseq : nullptr, t - STEPS);
  }

  // logits = hseq @ W_out^T + b_out  -> out[640:]
  gemm_wmma_bf16<true, false><<<dim3(63, 40), 256, 0, stream>>>(
      hseq, HID, W_out, HID, b_out, nullptr, 0, out + TB, VOCAB,
      TB, VOCAB, HID, 0);
  // argmax over vocab per row -> out[0:640] (as float indices)
  argmax_kernel<<<TB, 256, 0, stream>>>(out + TB, out);
}
